// RNN_1297080123709
// MI455X (gfx1250) — matrix operand order repaired and checked by reading, not compiled, not run on hardware
//
#include <hip/hip_runtime.h>

// ---------------------------------------------------------------------------
// LSTM char-RNN forward for MI455X (gfx1250, wave32, WMMA bf16).
// 64 blocks x 512 threads; block owns 16 batch rows for all 512 steps.
// Wave w owns hidden columns jc = w*16..+16 across ALL 4 gates (N-tiles
// {g*16+w}): LSTM cell update is pure register math, c state in VGPRs,
// one barrier per timestep. Wh streamed as bf16 WMMA fragments from L2.
// ---------------------------------------------------------------------------

#define TSEQ   512
#define BATCH  1024
#define UNITS  256
#define NC     128
#define G4     1024              // 4*UNITS
#define MB     16                // batch rows per block
#define NBLOCKS (BATCH / MB)     // 64
#define TPB    512
#define NWAVES (TPB / 32)        // 16

typedef __attribute__((ext_vector_type(16))) __bf16 v16bf;
typedef __attribute__((ext_vector_type(8)))  float  v8f;

union Frag {
    uint4 u[2];
    v16bf v;
};

__device__ __forceinline__ unsigned short f2bf(float x) {
    unsigned int u = __float_as_uint(x);
    unsigned int r = u + 0x7FFFu + ((u >> 16) & 1u);   // round-to-nearest-even
    return (unsigned short)(r >> 16);
}

__device__ __forceinline__ float fast_sigmoid(float z) {
    return 1.0f / (1.0f + __expf(-z));                 // saturates cleanly
}
__device__ __forceinline__ float stable_tanh(float x) {
    float ax = fabsf(x);
    float t  = 1.0f - 2.0f / (__expf(2.0f * ax) + 1.0f);   // exp->inf => t=1
    return copysignf(t, x);
}

// ---------------------------------------------------------------------------
// Kernel 0: weight prep.
//   whb: bf16 Wh, fragment-major. Fragment (nt,kt): K=kt*32..+32, N=nt*16..+16.
//        lane%16 = n_local, packed pos p -> k_local = 8*(lane/16)+(p&7)+16*(p>>3) -> two b128 loads per fragment.
//   wxb: f32 (Wx + b) transposed to [char][jc][gate] so one float4 load
//        fetches all 4 gate inputs for a given hidden column.
// ---------------------------------------------------------------------------
__global__ void prep_weights(const float* __restrict__ Wh,
                             const float* __restrict__ Wx,
                             const float* __restrict__ bvec,
                             unsigned short* __restrict__ whb,
                             float* __restrict__ wxb) {
    int tid = blockIdx.x * blockDim.x + threadIdx.x;
    if (tid < UNITS * G4) {
        int frag = tid >> 9;            // 512 elements per fragment
        int lane = (tid >> 4) & 31;
        int p    = tid & 15;
        int nt   = frag >> 3;
        int kt   = frag & 7;
        int k = kt * 32 + 8 * (lane >> 4) + (p & 7) + ((p >> 3) << 4);
        int n = nt * 16 + (lane & 15);
        whb[tid] = f2bf(Wh[k * G4 + n]);
    }
    if (tid < NC * G4) {
        int ci   = tid >> 10;           // char index
        int rest = tid & 1023;
        int jc   = rest >> 2;           // hidden column 0..255
        int gate = rest & 3;            // i,f,g,o
        int src  = gate * UNITS + jc;
        wxb[tid] = Wx[ci * G4 + src] + bvec[src];
    }
}

// ---------------------------------------------------------------------------
// Kernel 1: the LSTM scan + output head.
// ---------------------------------------------------------------------------
__global__ __launch_bounds__(TPB, 1)
void lstm_kernel(const int* __restrict__ seq,              // [BATCH][TSEQ]
                 const unsigned short* __restrict__ whb,   // bf16 fragments
                 const float* __restrict__ wxb,            // [char][jc][gate]
                 const float* __restrict__ Wd,             // [UNITS][NC]
                 const float* __restrict__ bd,             // [NC]
                 float* __restrict__ out) {                // [BATCH][NC]
    extern __shared__ char smem[];
    // LDS (32896 B): h bf16 A-fragments, double-buffered indices, final h, logits
    unsigned short* hfr  = (unsigned short*)smem;                    // 8 KB (8 kt x 512)
    int*            idxl = (int*)(smem + 8192);                      // 128 B (2 x 16)
    float*          hlin = (float*)(smem + 8192 + 128);              // 16 KB
    float*          lgts = (float*)(smem + 8192 + 128 + 16384);      // 8 KB

    const int tid  = threadIdx.x;
    const int wave = tid >> 5;          // 0..15 == hidden-column tile
    const int lane = tid & 31;
    const int row0 = blockIdx.x * MB;

    for (int i = tid; i < 8 * 512; i += TPB) hfr[i] = 0;
    if (tid < MB) idxl[tid] = seq[(row0 + tid) * TSEQ + 0];
    __syncthreads();

    const int laneM = 8 * (lane >> 4);   // C/D layout: lanes 16-31 -> M+8
    const int laneN = lane & 15;
    const int jc    = wave * 16 + laneN; // this lane's hidden column (0..255)

    // Precompute this lane's h-fragment store base (A-layout inversion; jc is
    // lane-constant, so only +r*16 varies):
    //   lanes 0-15: V0..3 -> K0..7, V4..7 -> K16..23 ; lanes 16-31: +8
    const int kl   = jc & 31, kt2 = jc >> 5;
    const int hi   = kl >> 4, win = kl & 15;
    const int grp  = win >> 3, rem = win & 7;
    const int vv   = hi * 4 + (rem >> 1), hh = rem & 1;
    const int hbase = kt2 * 512 + grp * 256 + vv * 2 + hh + laneM * 16;

    float creg[8];                      // c state, register-resident
    #pragma unroll
    for (int r = 0; r < 8; ++r) creg[r] = 0.0f;

    for (int t = 0; t < TSEQ; ++t) {
        const int* cur = idxl + (t & 1) * MB;
        if (t + 1 < TSEQ && tid < MB)
            idxl[((t + 1) & 1) * MB + tid] = seq[(row0 + tid) * TSEQ + (t + 1)];

        // ---- z = h @ Wh ; wave's N-tiles are {g*16 + wave}, g = gate ----
        v8f acc[4];
        #pragma unroll
        for (int g = 0; g < 4; ++g) acc[g] = v8f{};

        #pragma unroll 2
        for (int kt = 0; kt < 8; ++kt) {
            Frag a, bf[4];
            const uint4* pa = (const uint4*)(hfr + kt * 512 + lane * 16);
            a.u[0] = pa[0];
            a.u[1] = pa[1];
            #pragma unroll
            for (int g = 0; g < 4; ++g) {
                const uint4* p = (const uint4*)(whb + (((g * 16 + wave) * 8 + kt) << 9) + lane * 16);
                bf[g].u[0] = p[0];
                bf[g].u[1] = p[1];
            }
            #pragma unroll
            for (int g = 0; g < 4; ++g)
                acc[g] = __builtin_amdgcn_wmma_f32_16x16x32_bf16(
                    false, a.v, false, bf[g].v, (short)0, acc[g], false, false);
        }

        // ---- register-resident LSTM cell update ----
        const bool last = (t == TSEQ - 1);
        #pragma unroll
        for (int r = 0; r < 8; ++r) {
            const int m  = laneM + r;
            const int ci = cur[m];
            const float4 x = *(const float4*)(wxb + (size_t)ci * G4 + jc * 4);
            float iv = fast_sigmoid(acc[0][r] + x.x);
            float fv = fast_sigmoid(acc[1][r] + x.y);
            float gv = stable_tanh (acc[2][r] + x.z);
            float ov = fast_sigmoid(acc[3][r] + x.w);
            float cn = fv * creg[r] + iv * gv;
            creg[r] = cn;
            float hn = ov * stable_tanh(cn);
            hfr[hbase + r * 16] = f2bf(hn);
            if (last) hlin[m * UNITS + jc] = hn;
        }
        __syncthreads();   // h fragments + next indices ready
    }

    // ---- logits = h @ Wd + bd (one-off, f32) ----
    #pragma unroll
    for (int q = 0; q < (MB * NC) / TPB; ++q) {   // 4 per thread
        int e = q * TPB + tid;
        int m = e >> 7, n = e & 127;
        float s = bd[n];
        for (int k = 0; k < UNITS; ++k)
            s += hlin[m * UNITS + k] * Wd[k * NC + n];
        lgts[e] = s;
    }
    __syncthreads();

    // ---- softmax per row ----
    if (tid < MB) {
        const float* r = lgts + tid * NC;
        float mx = r[0];
        for (int n = 1; n < NC; ++n) mx = fmaxf(mx, r[n]);
        float s = 0.0f;
        for (int n = 0; n < NC; ++n) s += __expf(r[n] - mx);
        float inv = 1.0f / s;
        float* o = out + (size_t)(row0 + tid) * NC;
        for (int n = 0; n < NC; ++n) o[n] = __expf(r[n] - mx) * inv;
    }
}

// ---------------------------------------------------------------------------
extern "C" void kernel_launch(void* const* d_in, const int* in_sizes, int n_in,
                              void* d_out, int out_size, void* d_ws, size_t ws_size,
                              hipStream_t stream) {
    const int*   seq = (const int*)d_in[0];    // inputs [1024,512]
    const float* Wx  = (const float*)d_in[1];  // [128,1024]
    const float* Wh  = (const float*)d_in[2];  // [256,1024]
    const float* b   = (const float*)d_in[3];  // [1024]
    const float* Wd  = (const float*)d_in[4];  // [256,128]
    const float* bd  = (const float*)d_in[5];  // [128]
    float* out = (float*)d_out;

    // workspace: 512KB bf16 Wh fragments + 512KB f32 (Wx+b, gate-interleaved)
    unsigned short* whb = (unsigned short*)d_ws;
    float*          wxb = (float*)((char*)d_ws + (size_t)UNITS * G4 * sizeof(unsigned short));

    prep_weights<<<(UNITS * G4 + 255) / 256, 256, 0, stream>>>(Wh, Wx, b, whb, wxb);

    const size_t shmem = 8192 + 128 + 16384 + 8192; // 32896 B
    lstm_kernel<<<NBLOCKS, TPB, shmem, stream>>>(seq, whb, wxb, Wd, bd, out);
}
